// EsmFoldingTrunk_82824149336559
// MI455X (gfx1250) — matrix operand order repaired and checked by reading, not compiled, not run on hardware
//
#include <hip/hip_runtime.h>

// ---------------------------------------------------------------------------
// EsmFoldingTrunk block for MI455X (gfx1250), wave32 + WMMA bf16.
// All matrix math routed through v_wmma_f32_16x16x32_bf16.
// LDS tile fills use GLOBAL_LOAD_ASYNC_TO_LDS_B128 when the toolchain exposes
// the builtin; next-K-chunk global_prefetch_b8 issued each iteration.
// ---------------------------------------------------------------------------

#define L_SEQ 256
#define C_S 1024
#define C_Z 128
#define H_S 32
#define H_Z 4
#define INF_V 1e9f
#define LN_EPS 1e-5f

typedef unsigned short u16;
typedef unsigned int u32;
typedef unsigned long long u64;

typedef __bf16 bf16_t;
typedef bf16_t v16bf __attribute__((ext_vector_type(16)));
typedef float v8f __attribute__((ext_vector_type(8)));

#if __has_builtin(__builtin_amdgcn_global_load_async_to_lds_b128) && \
    __has_builtin(__builtin_amdgcn_s_wait_asynccnt)
#define USE_ASYNC_LDS 1
// builtin prototype (from clang diagnostics): (v4i AS1*, v4i AS3*, Ii, Ii)
typedef int v4i_vs __attribute__((vector_size(16)));
typedef __attribute__((address_space(1))) v4i_vs as1_v4i;
typedef __attribute__((address_space(3))) v4i_vs as3_v4i;
#define GLBP(p) ((as1_v4i*)(p))
#define LDSP(p) ((as3_v4i*)(p))
#else
#define USE_ASYNC_LDS 0
#endif

union Frag {
  v16bf v;
  uint4 u[2];
  u16 s[16];
};

// native f32 -> bf16 convert (backend picks v_cvt_*bf16* on gfx1250)
__device__ inline u16 f2bf(float f) {
  return __builtin_bit_cast(u16, (__bf16)f);
}
__device__ inline u32 f2bf2(float a, float b) {
  return (u32)f2bf(a) | ((u32)f2bf(b) << 16);
}
__device__ inline uint4 pk8(float4 a, float4 b) {
  uint4 r;
  r.x = f2bf2(a.x, a.y);
  r.y = f2bf2(a.z, a.w);
  r.z = f2bf2(b.x, b.y);
  r.w = f2bf2(b.z, b.w);
  return r;
}

__device__ inline float sigmoidf_(float x) { return 1.0f / (1.0f + __expf(-x)); }

__device__ inline v8f wmma_bf16(const Frag& a, const Frag& b, v8f c) {
  return __builtin_amdgcn_wmma_f32_16x16x32_bf16(false, a.v, false, b.v,
                                                 (short)0, c, false, false);
}

// A-matrix (16x32 bf16) per-lane element e -> K index (ISA 7.12.2):
//   c = 16*(e>>3) + 8*half + (e&7)   (two contiguous 8-wide runs)
// B-matrix (32x16 bf16) per-lane element e -> K index: same as A, c = 16*(e>>3) + 8*half + (e&7)

// ---------------------------------------------------------------------------
// Generic WMMA GEMM: C[M,N] = act(A[M,K] @ Bt[N,K]^T + bias) (+= C if accOut)
// A: f32 or bf16 row-major. Bt: bf16, row n holds K contiguous (i.e. B^T).
// Block: 256 threads (8 waves), tile 128x64, wave tile 32x32 (4 WMMA/K-step),
// K-step 32. Batched via grid.z.
// ---------------------------------------------------------------------------
__global__ __launch_bounds__(256) void gemm_wmma_k(
    const void* __restrict__ Ap, int aIsF32, u64 strideA,
    const u16* __restrict__ Bt, u64 strideB,
    float* __restrict__ C, u64 strideC,
    const float* __restrict__ bias, int act, int accOut,
    int M, int N, int K) {
  __shared__ alignas(16) u16 lA[128 * 32];
  __shared__ alignas(16) u16 lB[64 * 32];

  const int t = threadIdx.x;
  const int bz = blockIdx.z;
  const int m0 = blockIdx.y * 128;
  const int n0 = blockIdx.x * 64;

  const int ar = t >> 1;        // 0..127: A tile row loaded by this thread
  const int ac = (t & 1) * 16;  // 16-wide K chunk
  const int br = t >> 2;        // 0..63: B tile row
  const int bc = (t & 3) * 8;   // 8-wide K chunk

  const int wid = t >> 5;
  const int lane = t & 31;
  const int half = lane >> 4;
  const int mr = lane & 15;
  const int wm = (wid & 3) * 32;   // wave M offset in tile
  const int wn = (wid >> 2) * 32;  // wave N offset in tile

  const float* Af = (const float*)Ap + (u64)bz * strideA;
  const u16* Ab = (const u16*)Ap + (u64)bz * strideA;
  const u16* Bb = Bt + (u64)bz * strideB;
  float* Cb = C + (u64)bz * strideC;

  const int gm = m0 + ar;
  const int gn = n0 + br;

  v8f acc[2][2];
  acc[0][0] = {};
  acc[0][1] = {};
  acc[1][0] = {};
  acc[1][1] = {};

  for (int kb = 0; kb < K; kb += 32) {
    // ---- stage A tile ----
    if (gm < M) {
      if (aIsF32) {
        const float* src = Af + (u64)gm * K + kb + ac;
        float4 x0 = *(const float4*)(src + 0);
        float4 x1 = *(const float4*)(src + 4);
        float4 x2 = *(const float4*)(src + 8);
        float4 x3 = *(const float4*)(src + 12);
        *(uint4*)&lA[ar * 32 + ac] = pk8(x0, x1);
        *(uint4*)&lA[ar * 32 + ac + 8] = pk8(x2, x3);
        if (kb + 32 < K) __builtin_prefetch(src + 32, 0, 1);
      } else {
        const u16* src = Ab + (u64)gm * K + kb + ac;
#if USE_ASYNC_LDS
        __builtin_amdgcn_global_load_async_to_lds_b128(
            GLBP(src), LDSP(&lA[ar * 32 + ac]), 0, 0);
        __builtin_amdgcn_global_load_async_to_lds_b128(
            GLBP(src + 8), LDSP(&lA[ar * 32 + ac + 8]), 0, 0);
#else
        *(uint4*)&lA[ar * 32 + ac] = *(const uint4*)(src);
        *(uint4*)&lA[ar * 32 + ac + 8] = *(const uint4*)(src + 8);
#endif
        if (kb + 32 < K) __builtin_prefetch(src + 32, 0, 1);
      }
    } else {
      uint4 z;
      z.x = z.y = z.z = z.w = 0u;
      *(uint4*)&lA[ar * 32 + ac] = z;
      *(uint4*)&lA[ar * 32 + ac + 8] = z;
    }

    // ---- stage B tile ----
    if (gn < N) {
      const u16* src = Bb + (u64)gn * K + kb + bc;
#if USE_ASYNC_LDS
      __builtin_amdgcn_global_load_async_to_lds_b128(
          GLBP(src), LDSP(&lB[br * 32 + bc]), 0, 0);
#else
      *(uint4*)&lB[br * 32 + bc] = *(const uint4*)(src);
#endif
      if (kb + 32 < K) __builtin_prefetch(src + 32, 0, 1);
    } else {
      uint4 z;
      z.x = z.y = z.z = z.w = 0u;
      *(uint4*)&lB[br * 32 + bc] = z;
    }

#if USE_ASYNC_LDS
    __builtin_amdgcn_s_wait_asynccnt(0);
#endif
    __syncthreads();

    Frag fa0, fa1, fb0, fb1;
    // A frags: interleaved K layout
    fa0.u[0] = *(const uint4*)&lA[(wm + mr) * 32 + 8 * half];
    fa0.u[1] = *(const uint4*)&lA[(wm + mr) * 32 + 16 + 8 * half];
    fa1.u[0] = *(const uint4*)&lA[(wm + 16 + mr) * 32 + 8 * half];
    fa1.u[1] = *(const uint4*)&lA[(wm + 16 + mr) * 32 + 16 + 8 * half];
    // B frags: interleaved K layout (same as A)
    fb0.u[0] = *(const uint4*)&lB[(wn + mr) * 32 + 8 * half];
    fb0.u[1] = *(const uint4*)&lB[(wn + mr) * 32 + 16 + 8 * half];
    fb1.u[0] = *(const uint4*)&lB[(wn + 16 + mr) * 32 + 8 * half];
    fb1.u[1] = *(const uint4*)&lB[(wn + 16 + mr) * 32 + 16 + 8 * half];

    acc[0][0] = wmma_bf16(fa0, fb0, acc[0][0]);
    acc[0][1] = wmma_bf16(fa0, fb1, acc[0][1]);
    acc[1][0] = wmma_bf16(fa1, fb0, acc[1][0]);
    acc[1][1] = wmma_bf16(fa1, fb1, acc[1][1]);

    __syncthreads();
  }

  // Epilogue: D layout lane(n)=mr, row m = 8*half + r per VGPR r.
  for (int ai = 0; ai < 2; ++ai) {
    for (int bi = 0; bi < 2; ++bi) {
      const int nn = n0 + wn + bi * 16 + mr;
      if (nn >= N) continue;
      const float bv = bias ? bias[nn] : 0.0f;
      for (int r = 0; r < 8; ++r) {
        const int m = m0 + wm + ai * 16 + 8 * half + r;
        if (m >= M) continue;
        float v = acc[ai][bi][r] + bv;
        if (act == 1) v = fmaxf(v, 0.0f);
        else if (act == 2) v = sigmoidf_(v);
        const u64 ci = (u64)m * N + nn;
        if (accOut) v += Cb[ci];
        Cb[ci] = v;
      }
    }
  }
}

// ---------------------------------------------------------------------------
// Fused attention (head_dim = 32 = one WMMA K-step), 1 wave per block.
// Task = (outer oi, head h, 16-query tile jt).  Keys = L_SEQ.
//   logits = (q*qscale) @ k^T + bias[(q*L+k)*biasStride + h]
//            + (rowm*maskf[k]-1)*INF ; softmax ; O = P @ V
// ---------------------------------------------------------------------------
__global__ __launch_bounds__(32) void attn_wmma_k(
    const float* __restrict__ Q, const float* __restrict__ Kx,
    const float* __restrict__ V, float* __restrict__ O,
    u64 oStrideQ, int pStrideQ, int hStrideQ,
    u64 oStrideO, int pStrideO, int hStrideO,
    const float* __restrict__ bias, int biasStride,
    const float* __restrict__ maskf, int useRowMask, float qscale) {
  __shared__ alignas(16) float sL[16 * L_SEQ];  // logits / probs f32
  __shared__ alignas(16) u16 sP[16 * L_SEQ];    // probs bf16

  const int jt = blockIdx.x;
  const int h = blockIdx.y;
  const int oi = blockIdx.z;
  const int lane = threadIdx.x;
  const int half = lane >> 4;
  const int mr = lane & 15;

  const float* qb = Q + oStrideQ * oi + h * hStrideQ;
  const float* kb = Kx + oStrideQ * oi + h * hStrideQ;
  const float* vb = V + oStrideQ * oi + h * hStrideQ;
  float* ob = O + oStrideO * oi + h * hStrideO;

  const float rowm = useRowMask ? maskf[oi] : 1.0f;

  // Q fragment (A layout): two contiguous 8-float runs per lane
  Frag fq;
  {
    const float* qr = qb + (u64)(jt * 16 + mr) * pStrideQ + 8 * half;
    float4 a0 = *(const float4*)(qr + 0);
    float4 a1 = *(const float4*)(qr + 4);
    float4 a2 = *(const float4*)(qr + 16);
    float4 a3 = *(const float4*)(qr + 20);
    a0.x *= qscale; a0.y *= qscale; a0.z *= qscale; a0.w *= qscale;
    a1.x *= qscale; a1.y *= qscale; a1.z *= qscale; a1.w *= qscale;
    a2.x *= qscale; a2.y *= qscale; a2.z *= qscale; a2.w *= qscale;
    a3.x *= qscale; a3.y *= qscale; a3.z *= qscale; a3.w *= qscale;
    fq.u[0] = pk8(a0, a1);
    fq.u[1] = pk8(a2, a3);
  }

  // logits, one WMMA per 16-key tile (K = head dim = 32)
  for (int kt = 0; kt < L_SEQ / 16; ++kt) {
    Frag fk;  // B layout: two contiguous 8-float runs per lane (same as A)
    const float* kr = kb + (u64)(kt * 16 + mr) * pStrideQ + 8 * half;
    float4 k0 = *(const float4*)(kr + 0);
    float4 k1 = *(const float4*)(kr + 4);
    float4 k2 = *(const float4*)(kr + 16);
    float4 k3 = *(const float4*)(kr + 20);
    fk.u[0] = pk8(k0, k1);
    fk.u[1] = pk8(k2, k3);
    v8f lg = {};
    lg = wmma_bf16(fq, fk, lg);
    const int kpos = kt * 16 + mr;
    const float mk = (rowm * maskf[kpos] - 1.0f) * INF_V;
    for (int r = 0; r < 8; ++r) {
      const int mq = 8 * half + r;
      float v = lg[r] + mk;
      if (bias) v += bias[((u64)(jt * 16 + mq) * L_SEQ + kpos) * biasStride + h];
      sL[mq * L_SEQ + kpos] = v;
    }
  }
  __syncthreads();

  // softmax: lanes (mr, mr+16) cooperate on row mr
  {
    const int row = mr;
    const int c0 = half * (L_SEQ / 2);
    float mx = -3.4e38f;
    for (int c = 0; c < L_SEQ / 2; ++c) mx = fmaxf(mx, sL[row * L_SEQ + c0 + c]);
    mx = fmaxf(mx, __shfl_xor(mx, 16, 32));
    float s = 0.0f;
    for (int c = 0; c < L_SEQ / 2; ++c) {
      const float e = __expf(sL[row * L_SEQ + c0 + c] - mx);
      sL[row * L_SEQ + c0 + c] = e;
      s += e;
    }
    s += __shfl_xor(s, 16, 32);
    const float inv = 1.0f / s;
    for (int c = 0; c < L_SEQ / 2; c += 2) {
      const u32 p = f2bf2(sL[row * L_SEQ + c0 + c] * inv,
                          sL[row * L_SEQ + c0 + c + 1] * inv);
      *(u32*)&sP[row * L_SEQ + c0 + c] = p;
    }
  }
  __syncthreads();

  // O[16,32] = P[16,L] @ V[L,32] : WMMA over 32-key chunks, 2 column frags
  v8f oa0 = {};
  v8f oa1 = {};
  for (int kc = 0; kc < L_SEQ / 32; ++kc) {
    Frag fp;
    fp.u[0] = *(const uint4*)&sP[mr * L_SEQ + kc * 32 + 8 * half];
    fp.u[1] = *(const uint4*)&sP[mr * L_SEQ + kc * 32 + 16 + 8 * half];
    Frag fv0, fv1;  // V^T gather (transpose): scalar column loads
    for (int e = 0; e < 16; ++e) {
      const int kk = kc * 32 + (e < 8 ? 8 * half + e : 16 + 8 * half + (e - 8));
      const float* vr = vb + (u64)kk * pStrideQ;
      fv0.s[e] = f2bf(vr[mr]);
      fv1.s[e] = f2bf(vr[mr + 16]);
    }
    oa0 = wmma_bf16(fp, fv0, oa0);
    oa1 = wmma_bf16(fp, fv1, oa1);
  }
  for (int r = 0; r < 8; ++r) {
    const int qg = jt * 16 + 8 * half + r;
    float* orow = ob + (u64)qg * pStrideO;
    orow[mr] = oa0[r];
    orow[mr + 16] = oa1[r];
  }
}

// ---------------------------------------------------------------------------
// LayerNorm over last dim, one wave per row (wave32 shuffle reduction).
// ---------------------------------------------------------------------------
__global__ __launch_bounds__(256) void layernorm_k(
    const float* __restrict__ x, const float* __restrict__ gg,
    const float* __restrict__ bb, float* __restrict__ y, long long rows, int D) {
  const int wid = threadIdx.x >> 5;
  const int lane = threadIdx.x & 31;
  const long long row = (long long)blockIdx.x * 8 + wid;
  if (row >= rows) return;
  const float* xr = x + row * D;
  float* yr = y + row * D;
  float s = 0.0f, s2 = 0.0f;
  for (int d = lane; d < D; d += 32) {
    const float v = xr[d];
    s += v;
    s2 += v * v;
  }
  for (int m = 16; m; m >>= 1) {
    s += __shfl_xor(s, m, 32);
    s2 += __shfl_xor(s2, m, 32);
  }
  const float mu = s / D;
  const float var = s2 / D - mu * mu;
  const float rs = rsqrtf(var + LN_EPS);
  for (int d = lane; d < D; d += 32) yr[d] = (xr[d] - mu) * rs * gg[d] + bb[d];
}

// --------------------------- elementwise helpers ---------------------------
__global__ void maskf_k(const unsigned char* m, float* mf, int n) {
  const int i = blockIdx.x * 256 + threadIdx.x;
  if (i < n) mf[i] = m[i] ? 1.0f : 0.0f;
}

__global__ void mul_sig_k(float* o, const float* g, long long n) {
  for (long long i = (long long)blockIdx.x * 256 + threadIdx.x; i < n;
       i += (long long)gridDim.x * 256)
    o[i] *= sigmoidf_(g[i]);
}

__global__ void add_gated_k(float* out, const float* t, const float* g, long long n) {
  for (long long i = (long long)blockIdx.x * 256 + threadIdx.x; i < n;
       i += (long long)gridDim.x * 256)
    out[i] += t[i] * sigmoidf_(g[i]);
}

// dst = mf[i]*mf[j] * sigmoid(gsrc) * psrc  over [L,L,C_Z]
__global__ void ab_mask_k(const float* gsrc, const float* psrc, const float* mf,
                          float* dst, long long n) {
  for (long long i = (long long)blockIdx.x * 256 + threadIdx.x; i < n;
       i += (long long)gridDim.x * 256) {
    const long long row = i >> 7;
    const int jj = (int)(row & (L_SEQ - 1));
    const int ii = (int)(row >> 8);
    dst[i] = mf[ii] * mf[jj] * sigmoidf_(gsrc[i]) * psrc[i];
  }
}

// bf16 weight pack: wt[N,K] = w[K,N]^T
__global__ void pack_w_k(const float* w, u16* wt, int K, int N) {
  const long long n = (long long)K * N;
  for (long long i = (long long)blockIdx.x * 256 + threadIdx.x; i < n;
       i += (long long)gridDim.x * 256) {
    const long long kk = i / N;
    const long long nn = i % N;
    wt[nn * K + kk] = f2bf(w[i]);
  }
}

// channel-major pack for triangle multiply:
// dst[c][i][k] = src[(trans ? (k,i) : (i,k))][c]  (bf16)
__global__ void pack_pair_k(const float* src, u16* dst, int trans) {
  const long long n = (long long)C_Z * L_SEQ * L_SEQ;
  for (long long i = (long long)blockIdx.x * 256 + threadIdx.x; i < n;
       i += (long long)gridDim.x * 256) {
    const int k = (int)(i & (L_SEQ - 1));
    const long long r = i >> 8;
    const int ii = (int)(r & (L_SEQ - 1));
    const int c = (int)(r >> 8);
    const long long s = trans ? ((long long)k * L_SEQ + ii) : ((long long)ii * L_SEQ + k);
    dst[i] = f2bf(src[s * C_Z + c]);
  }
}

// dst[(i*L+j)*C + c] = src[(c*L+i)*L + j]
__global__ void unpack_cij_k(const float* src, float* dst, long long n) {
  for (long long i = (long long)blockIdx.x * 256 + threadIdx.x; i < n;
       i += (long long)gridDim.x * 256) {
    const int c = (int)(i & (C_Z - 1));
    const long long r = i >> 7;
    const int jj = (int)(r & (L_SEQ - 1));
    const int ii = (int)(r >> 8);
    dst[i] = src[((long long)c * L_SEQ + ii) * L_SEQ + jj];
  }
}

// dst[i,j,c] = src[j,i,c]
__global__ void transpose_k(const float* src, float* dst, long long n) {
  for (long long i = (long long)blockIdx.x * 256 + threadIdx.x; i < n;
       i += (long long)gridDim.x * 256) {
    const int c = (int)(i & (C_Z - 1));
    const long long r = i >> 7;
    const int jj = (int)(r & (L_SEQ - 1));
    const int ii = (int)(r >> 8);
    dst[i] = src[((long long)jj * L_SEQ + ii) * C_Z + c];
  }
}

// out[i,j,c] += x[j,i,c]
__global__ void add_transposed_k(float* out, const float* x, long long n) {
  for (long long i = (long long)blockIdx.x * 256 + threadIdx.x; i < n;
       i += (long long)gridDim.x * 256) {
    const int c = (int)(i & (C_Z - 1));
    const long long r = i >> 7;
    const int jj = (int)(r & (L_SEQ - 1));
    const int ii = (int)(r >> 8);
    out[i] += x[((long long)jj * L_SEQ + ii) * C_Z + c];
  }
}

// pu[i,j,c] : c<64 -> qh[j,c]*kh[i,c] ; else qh[j,c-64] - kh[i,c-64]
__global__ void build_pu_k(const float* sn, float* dst, long long n) {
  for (long long i = (long long)blockIdx.x * 256 + threadIdx.x; i < n;
       i += (long long)gridDim.x * 256) {
    const int c = (int)(i & (C_Z - 1));
    const long long r = i >> 7;
    const int jj = (int)(r & (L_SEQ - 1));
    const int ii = (int)(r >> 8);
    float v;
    if (c < 64)
      v = sn[(long long)jj * C_Z + c] * sn[(long long)ii * C_Z + 64 + c];
    else
      v = sn[(long long)jj * C_Z + (c - 64)] - sn[(long long)ii * C_Z + c];
    dst[i] = v;
  }
}

// ---------------------------------------------------------------------------
// Host orchestration.  d_in order (jax tree-flatten, sorted keys in params):
//  0 sequence_state, 1 pairwise_state, 2 mask,
//  3 ln1_b, 4 ln1_g,
//  mlp_pair: 5 b1, 6 b2, 7 ln_b, 8 ln_g, 9 w1, 10 w2
//  mlp_seq : 11 b1,12 b2,13 ln_b,14 ln_g,15 w1,16 w2
//  p2s     : 17 ln_b,18 ln_g,19 w
//  s2p     : 20 bo,21 bp,22 ln_b,23 ln_g,24 wo,25 wp
//  sa      : 26 bg,27 bo,28 wg,29 wo,30 wqkv
//  tae     : 31 bg,32 bo,33 ln_b,34 ln_g,35 wg,36 wk,37 wo,38 wq,39 wtri,40 wv
//  tas     : 41..50 (same order)
//  tmi     : 51..66 (bag,bap,bbg,bbp,bg,bz,ln_in_b,ln_in_g,ln_out_b,ln_out_g,
//                    wag,wap,wbg,wbp,wg,wz)
//  tmo     : 67..82 (same order)
// ---------------------------------------------------------------------------
extern "C" void kernel_launch(void* const* d_in, const int* in_sizes, int n_in,
                              void* d_out, int out_size, void* d_ws, size_t ws_size,
                              hipStream_t stream) {
  (void)in_sizes; (void)n_in; (void)out_size; (void)ws_size;

  const float* in_seq = (const float*)d_in[0];
  const float* in_pair = (const float*)d_in[1];
  const unsigned char* in_mask = (const unsigned char*)d_in[2];
  auto P = [&](int i) { return (const float*)d_in[i]; };

  float* seq = (float*)d_out;                      // [256,1024]
  float* pair = (float*)d_out + (u64)L_SEQ * C_S;  // [256,256,128]

  const long long NP = (long long)L_SEQ * L_SEQ;
  const long long NPC = NP * C_Z;
  const long long NSC = (long long)L_SEQ * C_S;

  // ---- workspace arena ----
  char* cur = (char*)d_ws;
  auto alloc = [&](size_t bytes) {
    void* p = cur;
    cur += (bytes + 255) & ~(size_t)255;
    return p;
  };
  float* MASKF = (float*)alloc(L_SEQ * sizeof(float));
  float* TBb = (float*)alloc((size_t)NP * H_Z * sizeof(float));
  float* BS = (float*)alloc((size_t)NP * H_S * sizeof(float));
  float* Y = (float*)alloc((size_t)NSC * sizeof(float));
  float* QKV = (float*)alloc((size_t)L_SEQ * 3 * C_S * sizeof(float));
  float* G = (float*)alloc((size_t)NSC * sizeof(float));
  float* OB = (float*)alloc((size_t)NSC * sizeof(float));
  float* HSQ = (float*)alloc((size_t)L_SEQ * 4 * C_S * sizeof(float));
  float* SN = (float*)alloc((size_t)L_SEQ * C_Z * sizeof(float));
  float* S0 = (float*)alloc((size_t)NPC * sizeof(float));
  float* S1 = (float*)alloc((size_t)NPC * sizeof(float));
  float* S2 = (float*)alloc((size_t)NPC * sizeof(float));
  float* S3 = (float*)alloc((size_t)NPC * sizeof(float));
  float* S4 = (float*)alloc((size_t)NPC * sizeof(float));
  float* S5 = (float*)alloc((size_t)NPC * sizeof(float));

  auto blocks = [](long long n) { return dim3((unsigned)((n + 255) / 256)); };

  auto gemm = [&](const void* A, int aF32, u64 sA, const u16* Bt, u64 sB,
                  float* C, u64 sC, const float* bias, int act, int acc,
                  int M, int N, int K, int batch) {
    dim3 g((N + 63) / 64, (M + 127) / 128, batch);
    gemm_wmma_k<<<g, dim3(256), 0, stream>>>(A, aF32, sA, Bt, sB, C, sC,
                                             bias, act, acc, M, N, K);
  };
  auto ln = [&](const float* x, const float* g, const float* b, float* y,
                long long rows, int D) {
    layernorm_k<<<dim3((unsigned)((rows + 7) / 8)), 256, 0, stream>>>(x, g, b, y,
                                                                      rows, D);
  };
  auto packw = [&](int idx, int K, int N) {
    u16* dst = (u16*)alloc((size_t)K * N * sizeof(u16));
    pack_w_k<<<blocks((long long)K * N), 256, 0, stream>>>(P(idx), dst, K, N);
    return dst;
  };

  // ---- init running state in d_out ----
  (void)hipMemcpyAsync(seq, in_seq, (size_t)NSC * sizeof(float),
                       hipMemcpyDeviceToDevice, stream);
  (void)hipMemcpyAsync(pair, in_pair, (size_t)NPC * sizeof(float),
                       hipMemcpyDeviceToDevice, stream);
  maskf_k<<<dim3(1), 256, 0, stream>>>(in_mask, MASKF, L_SEQ);

  // ---- pack weights (f32 [K,N] -> bf16 [N,K]) ----
  u16* Wp2s = packw(19, C_Z, H_S);
  u16* Wqkv = packw(30, C_S, 3 * C_S);
  u16* Wsag = packw(28, C_S, C_S);
  u16* Wsao = packw(29, C_S, C_S);
  u16* Wm1s = packw(15, C_S, 4 * C_S);
  u16* Wm2s = packw(16, 4 * C_S, C_S);
  u16* Ws2pp = packw(25, C_S, C_Z);
  u16* Ws2po = packw(24, C_Z, C_Z);
  u16* Wm1p = packw(9, C_Z, 4 * C_Z);
  u16* Wm2p = packw(10, 4 * C_Z, C_Z);
  u16* Wtm[2][6];  // [tmo,tmi] x [wag,wap,wbg,wbp,wg,wz]
  const int tmBase[2] = {67, 51};
  for (int i = 0; i < 2; ++i)
    for (int j = 0; j < 6; ++j) Wtm[i][j] = packw(tmBase[i] + 10 + j, C_Z, C_Z);
  u16* Wta[2][6];  // [tas,tae] x [wq,wk,wv,wg,wo,wtri]
  const int taBase[2] = {41, 31};
  for (int i = 0; i < 2; ++i) {
    Wta[i][0] = packw(taBase[i] + 7, C_Z, C_Z);  // wq
    Wta[i][1] = packw(taBase[i] + 5, C_Z, C_Z);  // wk
    Wta[i][2] = packw(taBase[i] + 9, C_Z, C_Z);  // wv
    Wta[i][3] = packw(taBase[i] + 4, C_Z, C_Z);  // wg
    Wta[i][4] = packw(taBase[i] + 6, C_Z, C_Z);  // wo
    Wta[i][5] = packw(taBase[i] + 8, C_Z, H_Z);  // wtri
  }

  const float qscale = 0.17677669529663689f;  // 32^-0.5

  // ---- pair -> sequence attention bias (uses ORIGINAL pair) ----
  ln(pair, P(18), P(17), S0, NP, C_Z);
  gemm(S0, 1, 0, Wp2s, 0, BS, 0, nullptr, 0, 0, (int)NP, H_S, C_Z, 1);

  // ---- gated sequence self-attention ----
  ln(seq, P(4), P(3), Y, L_SEQ, C_S);
  gemm(Y, 1, 0, Wqkv, 0, QKV, 0, nullptr, 0, 0, L_SEQ, 3 * C_S, C_S, 1);
  attn_wmma_k<<<dim3(L_SEQ / 16, H_S, 1), 32, 0, stream>>>(
      QKV, QKV + 32, QKV + 64, OB,
      0ull, 3 * C_S, 96, 0ull, C_S, 32,
      BS, H_S, MASKF, 0, qscale);
  gemm(Y, 1, 0, Wsag, 0, G, 0, P(26), 0, 0, L_SEQ, C_S, C_S, 1);
  mul_sig_k<<<blocks(NSC), 256, 0, stream>>>(OB, G, NSC);
  gemm(OB, 1, 0, Wsao, 0, seq, 0, P(27), 0, 1, L_SEQ, C_S, C_S, 1);  // residual

  // ---- sequence MLP ----
  ln(seq, P(14), P(13), Y, L_SEQ, C_S);
  gemm(Y, 1, 0, Wm1s, 0, HSQ, 0, P(11), 1, 0, L_SEQ, 4 * C_S, C_S, 1);
  gemm(HSQ, 1, 0, Wm2s, 0, seq, 0, P(12), 0, 1, L_SEQ, C_S, 4 * C_S, 1);

  // ---- sequence -> pair (outer product / difference) ----
  ln(seq, P(23), P(22), Y, L_SEQ, C_S);
  gemm(Y, 1, 0, Ws2pp, 0, SN, 0, P(21), 0, 0, L_SEQ, C_Z, C_S, 1);
  build_pu_k<<<blocks(NPC), 256, 0, stream>>>(SN, S1, NPC);
  gemm(S1, 1, 0, Ws2po, 0, pair, 0, P(20), 0, 1, (int)NP, C_Z, C_Z, 1);

  // ---- triangle multiplicative updates ----
  auto tri_mul = [&](int base, int outgoing, u16* const* W) {
    ln(pair, P(base + 7), P(base + 6), S0, NP, C_Z);  // zn
    gemm(S0, 1, 0, W[0], 0, S1, 0, P(base + 0), 0, 0, (int)NP, C_Z, C_Z, 1);
    gemm(S0, 1, 0, W[1], 0, S2, 0, P(base + 1), 0, 0, (int)NP, C_Z, C_Z, 1);
    ab_mask_k<<<blocks(NPC), 256, 0, stream>>>(S1, S2, MASKF, S1, NPC);  // a
    gemm(S0, 1, 0, W[2], 0, S2, 0, P(base + 2), 0, 0, (int)NP, C_Z, C_Z, 1);
    gemm(S0, 1, 0, W[3], 0, S3, 0, P(base + 3), 0, 0, (int)NP, C_Z, C_Z, 1);
    ab_mask_k<<<blocks(NPC), 256, 0, stream>>>(S2, S3, MASKF, S2, NPC);  // b
    u16* PA = (u16*)S3;
    u16* PB = (u16*)S4;
    pack_pair_k<<<blocks(NPC), 256, 0, stream>>>(S1, PA, outgoing ? 0 : 1);
    pack_pair_k<<<blocks(NPC), 256, 0, stream>>>(S2, PB, outgoing ? 0 : 1);
    // 128-way batched 256x256x256 GEMM-NT (per-channel triangle einsum)
    gemm(PA, 0, (u64)NP, PB, (u64)NP, S5, (u64)NP, nullptr, 0, 0,
         L_SEQ, L_SEQ, L_SEQ, C_Z);
    unpack_cij_k<<<blocks(NPC), 256, 0, stream>>>(S5, S1, NPC);
    ln(S1, P(base + 9), P(base + 8), S2, NP, C_Z);
    gemm(S2, 1, 0, W[5], 0, S1, 0, P(base + 5), 0, 0, (int)NP, C_Z, C_Z, 1);
    gemm(S0, 1, 0, W[4], 0, S2, 0, P(base + 4), 0, 0, (int)NP, C_Z, C_Z, 1);
    add_gated_k<<<blocks(NPC), 256, 0, stream>>>(pair, S1, S2, NPC);
  };
  tri_mul(67, 1, Wtm[0]);  // outgoing
  tri_mul(51, 0, Wtm[1]);  // incoming

  // ---- triangle attention (starting / ending node) ----
  auto tri_att = [&](int base, u16* const* W, int transposed) {
    const float* zin = pair;
    if (transposed) {
      transpose_k<<<blocks(NPC), 256, 0, stream>>>(pair, S0, NPC);
      zin = S0;
    }
    float* ZN = transposed ? S1 : S0;
    float* Qb = transposed ? S2 : S1;
    float* Kb = transposed ? S3 : S2;
    float* Vb = transposed ? S4 : S3;
    float* Ob = transposed ? S5 : S4;
    float* Gb = transposed ? S0 : S1;
    ln(zin, P(base + 3), P(base + 2), ZN, NP, C_Z);
    gemm(ZN, 1, 0, W[5], 0, TBb, 0, nullptr, 0, 0, (int)NP, H_Z, C_Z, 1);
    gemm(ZN, 1, 0, W[0], 0, Qb, 0, nullptr, 0, 0, (int)NP, C_Z, C_Z, 1);
    gemm(ZN, 1, 0, W[1], 0, Kb, 0, nullptr, 0, 0, (int)NP, C_Z, C_Z, 1);
    gemm(ZN, 1, 0, W[2], 0, Vb, 0, nullptr, 0, 0, (int)NP, C_Z, C_Z, 1);
    attn_wmma_k<<<dim3(L_SEQ / 16, H_Z, L_SEQ), 32, 0, stream>>>(
        Qb, Kb, Vb, Ob,
        (u64)L_SEQ * C_Z, C_Z, 32, (u64)L_SEQ * C_Z, C_Z, 32,
        TBb, H_Z, MASKF, 1, qscale);
    gemm(ZN, 1, 0, W[3], 0, Gb, 0, P(base + 0), 0, 0, (int)NP, C_Z, C_Z, 1);
    mul_sig_k<<<blocks(NPC), 256, 0, stream>>>(Ob, Gb, NPC);
    if (!transposed) {
      gemm(Ob, 1, 0, W[4], 0, pair, 0, P(base + 1), 0, 1, (int)NP, C_Z, C_Z, 1);
    } else {
      float* T = Qb;  // free after attention
      gemm(Ob, 1, 0, W[4], 0, T, 0, P(base + 1), 0, 0, (int)NP, C_Z, C_Z, 1);
      add_transposed_k<<<blocks(NPC), 256, 0, stream>>>(pair, T, NPC);
    }
  };
  tri_att(41, Wta[0], 0);  // tas
  tri_att(31, Wta[1], 1);  // tae (on transposed pair, residual re-transposed)

  // ---- pair MLP (chunked: hidden would otherwise be 134MB) ----
  ln(pair, P(8), P(7), S0, NP, C_Z);
  for (int ch = 0; ch < 4; ++ch) {
    const long long r0 = (long long)ch * (NP / 4);
    gemm(S0 + r0 * C_Z, 1, 0, Wm1p, 0, S1, 0, P(5), 1, 0,
         (int)(NP / 4), 4 * C_Z, C_Z, 1);
    gemm(S1, 1, 0, Wm2p, 0, pair + r0 * C_Z, 0, P(6), 0, 1,
         (int)(NP / 4), C_Z, 4 * C_Z, 1);
  }
}
